// CDC_62646392980082
// MI455X (gfx1250) — matrix operand order repaired and checked by reading, not compiled, not run on hardware
//
#include <hip/hip_runtime.h>
#include <hip/hip_bf16.h>
#include <stdint.h>

// ---------------------------------------------------------------------------
// CDNA5 (gfx1250) wave32 WMMA types / helpers
// ---------------------------------------------------------------------------
typedef __attribute__((ext_vector_type(16))) __bf16 v16bf;
typedef __attribute__((ext_vector_type(8)))  __bf16 v8bf;
typedef __attribute__((ext_vector_type(8)))  float  v8f;

__device__ __forceinline__ __bf16 f2bf(float f) {
  unsigned int u = __float_as_uint(f);
  u += 0x7FFFu + ((u >> 16) & 1u);          // round-to-nearest-even
  unsigned short s = (unsigned short)(u >> 16);
  return __builtin_bit_cast(__bf16, s);
}

__device__ __forceinline__ v8f wmma_bf16(v16bf a, v16bf b, v8f c) {
  // D = A(16x32 bf16) * B(32x16 bf16) + C(16x16 f32)
  return __builtin_amdgcn_wmma_f32_16x16x32_bf16(
      /*neg_a=*/false, a, /*neg_b=*/false, b,
      /*c_mod=*/(short)0, c, /*reuse_a=*/false, /*reuse_b=*/false);
}

// A-matrix (16x32 bf16): lane = m + 16*hi ; element e -> K = 16*(e/8) + 8*hi + (e&7)
// Two aligned 16B loads per lane.
__device__ __forceinline__ v16bf loadA_bf(const __bf16* __restrict__ row, int kk, int hi) {
  v8bf lo = *(const v8bf*)(row + kk + hi * 8);
  v8bf h8 = *(const v8bf*)(row + kk + hi * 8 + 16);
  return __builtin_shufflevector(lo, h8, 0, 1, 2, 3, 4, 5, 6, 7,
                                         8, 9, 10, 11, 12, 13, 14, 15);
}
// B-matrix (32x16 bf16): lane = n + 16*hi ; element e -> K = 16*(e/8) + 8*hi + (e&7)
__device__ __forceinline__ v16bf loadB_bf(const __bf16* __restrict__ col, int kk, int hi) {
  v8bf lo = *(const v8bf*)(col + kk + hi * 8);
  v8bf h8 = *(const v8bf*)(col + kk + hi * 8 + 16);
  return __builtin_shufflevector(lo, h8, 0, 1, 2, 3, 4, 5, 6, 7,
                                         8, 9, 10, 11, 12, 13, 14, 15);
}

// Problem constants
#define PB 256
#define PK 5
#define PR 6
#define PC 7
#define PP 1280
#define PH 256
#define PS 64
#define BC 1792          // B*C
#define MROWS 10752      // R*B*C
#define CHUNK 32         // b-chunk for preds/dots
#define GROUPS (CHUNK * PK)          // 160
#define NPREDS 35840.0f  // sum(mask)*B*C = 20*256*7

// ---------------------------------------------------------------------------
// f32 -> bf16 bulk conversion (packed 2-per-thread, uint stores)
// ---------------------------------------------------------------------------
__global__ void k_cvt(const float* __restrict__ src, unsigned int* __restrict__ dst,
                      int n2) {
  int i = blockIdx.x * blockDim.x + threadIdx.x;
  if (i < n2) {
    float2 v = ((const float2*)src)[i];
    unsigned int u0 = __float_as_uint(v.x); u0 += 0x7FFFu + ((u0 >> 16) & 1u);
    unsigned int u1 = __float_as_uint(v.y); u1 += 0x7FFFu + ((u1 >> 16) & 1u);
    dst[i] = (u0 >> 16) | (u1 & 0xFFFF0000u);
  }
}

// ---------------------------------------------------------------------------
// init: zero accumulators, broadcast hidden into h0 (bf16)
// ---------------------------------------------------------------------------
__global__ void k_init(__bf16* __restrict__ h0, const float* __restrict__ hidden,
                       float* __restrict__ accbuf) {
  int idx = blockIdx.x * blockDim.x + threadIdx.x;
  if (idx < 2) accbuf[idx] = 0.0f;
  if (idx < BC * PH) h0[idx] = f2bf(hidden[idx & (PH - 1)]);
}

// ---------------------------------------------------------------------------
// GI = X @ W_ih^T + b_ih ; X row m=(r*1792 + b*7 + c) is encodings[b,r,c,:]
// M=10752, N=768, K=1280 ; each wave does one M-tile x two N-tiles.
// ---------------------------------------------------------------------------
__global__ __launch_bounds__(128) void k_gi_gemm(const __bf16* __restrict__ encb,
                                                 const __bf16* __restrict__ Wb,
                                                 const float* __restrict__ b_ih,
                                                 float* __restrict__ GI) {
  const int lane = threadIdx.x & 31, hi = lane >> 4, lo = lane & 15;
  int tile = blockIdx.x * 4 + (threadIdx.x >> 5);
  if (tile >= (MROWS / 16) * (768 / 32)) return;
  const int np = tile % 24, mt = tile / 24;

  const int m = mt * 16 + lo;
  const int r = m / BC, t = m % BC, b = t / PC, c = t % PC;
  const __bf16* arow = encb + (size_t)(b * 49 + r * 7 + c) * PP;
  const int n0 = np * 32 + lo;
  const __bf16* bc0 = Wb + (size_t)n0 * PP;
  const __bf16* bc1 = Wb + (size_t)(n0 + 16) * PP;

  v8f acc0 = {}, acc1 = {};
  for (int kk = 0; kk < PP; kk += 32) {
    __builtin_prefetch(arow + kk + 128, 0, 3);
    v16bf a = loadA_bf(arow, kk, hi);
    acc0 = wmma_bf16(a, loadB_bf(bc0, kk, hi), acc0);
    acc1 = wmma_bf16(a, loadB_bf(bc1, kk, hi), acc1);
  }

  const float bias0 = b_ih[n0], bias1 = b_ih[n0 + 16];
#pragma unroll
  for (int i = 0; i < 8; ++i) {
    int row = mt * 16 + i + hi * 8;
    GI[(size_t)row * 768 + n0]      = acc0[i] + bias0;
    GI[(size_t)row * 768 + n0 + 16] = acc1[i] + bias1;
  }
}

// ---------------------------------------------------------------------------
// One GRU step: GH tiles for r/z/n gates (shared A = h_in) + gate math.
// M=1792, N=256 (x3 gates), K=256 ; h kept in bf16 (ping-pong).
// ---------------------------------------------------------------------------
__global__ __launch_bounds__(128) void k_gru_step(const float* __restrict__ GI,
                                                  const __bf16* __restrict__ Whhb,
                                                  const float* __restrict__ b_hh,
                                                  const __bf16* __restrict__ h_in,
                                                  __bf16* __restrict__ h_out,
                                                  __bf16* __restrict__ ctxb, int r) {
  const int lane = threadIdx.x & 31, hi = lane >> 4, lo = lane & 15;
  int tile = blockIdx.x * 4 + (threadIdx.x >> 5);
  if (tile >= (BC / 16) * (PH / 16)) return;
  const int jt = tile % 16, mt = tile / 16;

  const int m = mt * 16 + lo;
  const __bf16* arow = h_in + (size_t)m * PH;
  const int col = jt * 16 + lo;
  const __bf16* bc0 = Whhb + (size_t)(col) * PH;
  const __bf16* bc1 = Whhb + (size_t)(256 + col) * PH;
  const __bf16* bc2 = Whhb + (size_t)(512 + col) * PH;

  v8f a0 = {}, a1 = {}, a2 = {};
#pragma unroll
  for (int kk = 0; kk < PH; kk += 32) {
    v16bf a = loadA_bf(arow, kk, hi);
    a0 = wmma_bf16(a, loadB_bf(bc0, kk, hi), a0);
    a1 = wmma_bf16(a, loadB_bf(bc1, kk, hi), a1);
    a2 = wmma_bf16(a, loadB_bf(bc2, kk, hi), a2);
  }

  const float br = b_hh[col], bz = b_hh[256 + col], bn = b_hh[512 + col];
#pragma unroll
  for (int i = 0; i < 8; ++i) {
    int row = mt * 16 + i + hi * 8;
    const float* gi = GI + (size_t)(r * BC + row) * 768;
    float gir = gi[col], giz = gi[256 + col], gin = gi[512 + col];
    float ghr = a0[i] + br, ghz = a1[i] + bz, ghn = a2[i] + bn;
    float rg = 1.0f / (1.0f + __expf(-(gir + ghr)));
    float zg = 1.0f / (1.0f + __expf(-(giz + ghz)));
    float ng = tanhf(gin + rg * ghn);
    float hold = (float)h_in[(size_t)row * PH + col];
    float hnew = (1.0f - zg) * ng + zg * hold;
    __bf16 hb = f2bf(hnew);
    h_out[(size_t)row * PH + col] = hb;
    ctxb[(size_t)(r * BC + row) * PH + col] = hb;
  }
}

// ---------------------------------------------------------------------------
// preds (one b-chunk): group g=(bl,k); rows rr=r*7+c (42 valid, padded to 48)
// pred[g,rr,p] = clip(ctx[r,b,c,:] @ Wk_w[k,p,:] + Wk_b[k,p], -1, 1)  -> bf16
// Each wave: one row-tile x two P-tiles.
// ---------------------------------------------------------------------------
__global__ __launch_bounds__(128) void k_preds(const __bf16* __restrict__ ctxb,
                                               const __bf16* __restrict__ Wkb,
                                               const float* __restrict__ Wk_b,
                                               __bf16* __restrict__ pred, int b0) {
  const int lane = threadIdx.x & 31, hi = lane >> 4, lo = lane & 15;
  int tile = blockIdx.x * 4 + (threadIdx.x >> 5);
  if (tile >= GROUPS * 3 * (PP / 32)) return;
  const int pt = tile % 40; tile /= 40;
  const int rt = tile % 3;  const int g = tile / 3;
  const int k = g % PK, bl = g / PK, b = b0 + bl;

  int rr = rt * 16 + lo; int rrc = rr > 41 ? 41 : rr;
  const int r = rrc / 7, c = rrc % 7;
  const __bf16* arow = ctxb + (size_t)(r * BC + b * PC + c) * PH;
  const int p0 = pt * 32 + lo;
  const __bf16* bc0 = Wkb + ((size_t)k * PP + p0) * PH;
  const __bf16* bc1 = Wkb + ((size_t)k * PP + p0 + 16) * PH;

  v8f acc0 = {}, acc1 = {};
#pragma unroll
  for (int kk = 0; kk < PH; kk += 32) {
    v16bf a = loadA_bf(arow, kk, hi);
    acc0 = wmma_bf16(a, loadB_bf(bc0, kk, hi), acc0);
    acc1 = wmma_bf16(a, loadB_bf(bc1, kk, hi), acc1);
  }

  const float bias0 = Wk_b[k * PP + p0], bias1 = Wk_b[k * PP + p0 + 16];
#pragma unroll
  for (int i = 0; i < 8; ++i) {
    int row = rt * 16 + i + hi * 8;   // 0..47
    float v0 = fminf(1.0f, fmaxf(-1.0f, acc0[i] + bias0));
    float v1 = fminf(1.0f, fmaxf(-1.0f, acc1[i] + bias1));
    pred[((size_t)g * 48 + row) * PP + p0]      = f2bf(v0);
    pred[((size_t)g * 48 + row) * PP + p0 + 16] = f2bf(v1);
  }
}

// ---------------------------------------------------------------------------
// dots (one b-chunk): dots[b,k,rr,ij] = pred[g,rr,:] . enc[b,ij,:]
// M=42(pad48), N=49(pad64), K=1280 ; one wave does one row-tile x ALL 4 N-tiles.
// ---------------------------------------------------------------------------
__global__ __launch_bounds__(128) void k_dots(const __bf16* __restrict__ pred,
                                              const __bf16* __restrict__ encb,
                                              float* __restrict__ dots, int b0) {
  const int lane = threadIdx.x & 31, hi = lane >> 4, lo = lane & 15;
  int tile = blockIdx.x * 4 + (threadIdx.x >> 5);
  if (tile >= GROUPS * 3) return;
  const int rt = tile % 3; const int g = tile / 3;
  const int k = g % PK, bl = g / PK, b = b0 + bl;

  const __bf16* arow = pred + ((size_t)g * 48 + rt * 16 + lo) * PP;
  const __bf16* bcol[4];
#pragma unroll
  for (int nt = 0; nt < 4; ++nt) {
    int n = nt * 16 + lo;
    int ij = n > 48 ? 48 : n;
    bcol[nt] = encb + ((size_t)b * 49 + ij) * PP;
  }

  v8f acc[4] = {{}, {}, {}, {}};
  for (int kk = 0; kk < PP; kk += 32) {
    __builtin_prefetch(arow + kk + 128, 0, 3);
    v16bf a = loadA_bf(arow, kk, hi);
#pragma unroll
    for (int nt = 0; nt < 4; ++nt)
      acc[nt] = wmma_bf16(a, loadB_bf(bcol[nt], kk, hi), acc[nt]);
  }

#pragma unroll
  for (int nt = 0; nt < 4; ++nt) {
    int n = nt * 16 + lo;
    if (n < 49) {
#pragma unroll
      for (int i = 0; i < 8; ++i) {
        int rr = rt * 16 + i + hi * 8;
        if (rr < 42)
          dots[((size_t)(b * PK + k) * 42 + rr) * 49 + n] = acc[nt][i];
      }
    }
  }
}

// ---------------------------------------------------------------------------
// loss/acc: per (b,k,r,c) softmax over [pos, 63 negatives]; masked r < R-k.
// ---------------------------------------------------------------------------
__global__ __launch_bounds__(256) void k_loss(const float* __restrict__ dots,
                                              const int* __restrict__ neg_rows,
                                              const int* __restrict__ neg_cols,
                                              float* __restrict__ accbuf) {
  __shared__ float sl[256], sa[256];
  const int idx = blockIdx.x * 256 + threadIdx.x;
  float loss = 0.0f, corr = 0.0f;
  if (idx < PB * PK * PR * PC) {
    const int c = idx % 7, r = (idx / 7) % 6, k = (idx / 42) % 5;
    if (r < PR - k) {
      const float* row = dots + (size_t)idx * 49;
      const int* nr = neg_rows + (size_t)idx * (PS - 1);
      const int* nc = neg_cols + (size_t)idx * (PS - 1);
      const float v0 = row[r * 7 + c];
      float mx = v0, mxr = -1e30f;
      float vals[PS - 1];
#pragma unroll 1
      for (int j = 0; j < PS - 1; ++j) {
        float v = row[nr[j] * 7 + nc[j]];
        vals[j] = v;
        mx = fmaxf(mx, v);
        mxr = fmaxf(mxr, v);
      }
      float s = __expf(v0 - mx);
#pragma unroll 1
      for (int j = 0; j < PS - 1; ++j) s += __expf(vals[j] - mx);
      loss = -(v0 - mx - __logf(s));
      corr = (v0 >= mxr) ? 1.0f : 0.0f;   // argmax==0 (ties pick index 0)
    }
  }
  sl[threadIdx.x] = loss; sa[threadIdx.x] = corr;
  __syncthreads();
  for (int s = 128; s > 0; s >>= 1) {
    if ((int)threadIdx.x < s) {
      sl[threadIdx.x] += sl[threadIdx.x + s];
      sa[threadIdx.x] += sa[threadIdx.x + s];
    }
    __syncthreads();
  }
  if (threadIdx.x == 0) {
    atomicAdd(&accbuf[0], sl[0]);
    atomicAdd(&accbuf[1], sa[0]);
  }
}

__global__ void k_final(const float* __restrict__ accbuf, float* __restrict__ out) {
  if (threadIdx.x == 0) {
    out[0] = accbuf[0] / NPREDS;
    out[1] = accbuf[1] / NPREDS;
  }
}

// ---------------------------------------------------------------------------
// Host launcher
// ---------------------------------------------------------------------------
extern "C" void kernel_launch(void* const* d_in, const int* in_sizes, int n_in,
                              void* d_out, int out_size, void* d_ws, size_t ws_size,
                              hipStream_t stream) {
  (void)in_sizes; (void)n_in; (void)out_size; (void)ws_size;
  const float* enc    = (const float*)d_in[0];
  const float* hidden = (const float*)d_in[1];
  const float* W_ih   = (const float*)d_in[2];
  const float* W_hh   = (const float*)d_in[3];
  const float* b_ih   = (const float*)d_in[4];
  const float* b_hh   = (const float*)d_in[5];
  const float* Wk_w   = (const float*)d_in[6];
  const float* Wk_b   = (const float*)d_in[7];
  const int* neg_rows = (const int*)d_in[8];
  const int* neg_cols = (const int*)d_in[9];
  float* out = (float*)d_out;

  char* ws = (char*)d_ws;
  size_t off = 0;
  auto alloc = [&](size_t bytes) {
    char* p = ws + off;
    off += (bytes + 255) & ~(size_t)255;
    return p;
  };
  const size_t nEnc = (size_t)PB * 49 * PP;      // 16,056,320
  const size_t nWih = (size_t)768 * PP;          //    983,040
  const size_t nWhh = (size_t)768 * PH;          //    196,608
  const size_t nWkw = (size_t)PK * PP * PH;      //  1,638,400

  __bf16* ENCB = (__bf16*)alloc(nEnc * 2);                        // 32.1 MB
  __bf16* WIHB = (__bf16*)alloc(nWih * 2);                        //  2.0 MB
  __bf16* WHHB = (__bf16*)alloc(nWhh * 2);                        //  0.4 MB
  __bf16* WKWB = (__bf16*)alloc(nWkw * 2);                        //  3.3 MB
  float*  GI   = (float*) alloc((size_t)MROWS * 768 * 4);         // 33.0 MB
  __bf16* CTXB = (__bf16*)alloc((size_t)MROWS * PH * 2);          //  5.5 MB
  __bf16* HB0  = (__bf16*)alloc((size_t)BC * PH * 2);             //  0.9 MB
  __bf16* HB1  = (__bf16*)alloc((size_t)BC * PH * 2);             //  0.9 MB
  float*  DOTS = (float*) alloc((size_t)PB * PK * 42 * 49 * 4);   // 10.5 MB
  __bf16* PRED = (__bf16*)alloc((size_t)GROUPS * 48 * PP * 2);    // 19.7 MB
  float*  ACC  = (float*) alloc(2 * 4);

  // bulk f32 -> bf16 of all WMMA operand matrices
  k_cvt<<<(int)((nEnc / 2 + 255) / 256), 256, 0, stream>>>(enc,  (unsigned int*)ENCB, (int)(nEnc / 2));
  k_cvt<<<(int)((nWih / 2 + 255) / 256), 256, 0, stream>>>(W_ih, (unsigned int*)WIHB, (int)(nWih / 2));
  k_cvt<<<(int)((nWhh / 2 + 255) / 256), 256, 0, stream>>>(W_hh, (unsigned int*)WHHB, (int)(nWhh / 2));
  k_cvt<<<(int)((nWkw / 2 + 255) / 256), 256, 0, stream>>>(Wk_w, (unsigned int*)WKWB, (int)(nWkw / 2));

  // init h0 + accumulators
  k_init<<<(BC * PH + 255) / 256, 256, 0, stream>>>(HB0, hidden, ACC);

  // GI for all timesteps at once
  {
    int nt = (MROWS / 16) * (768 / 32);
    k_gi_gemm<<<(nt + 3) / 4, 128, 0, stream>>>(ENCB, WIHB, b_ih, GI);
  }

  // sequential GRU scan (ping-pong h)
  __bf16* hin = HB0; __bf16* hout = HB1;
  for (int r = 0; r < PR; ++r) {
    int nt = (BC / 16) * (PH / 16);
    k_gru_step<<<(nt + 3) / 4, 128, 0, stream>>>(GI, WHHB, b_hh, hin, hout, CTXB, r);
    __bf16* tmp = hin; hin = hout; hout = tmp;
  }

  // preds + dots, chunked over b to bound workspace
  for (int b0 = 0; b0 < PB; b0 += CHUNK) {
    int ntp = GROUPS * 3 * (PP / 32);
    k_preds<<<(ntp + 3) / 4, 128, 0, stream>>>(CTXB, WKWB, Wk_b, PRED, b0);
    int ntd = GROUPS * 3;
    k_dots<<<(ntd + 3) / 4, 128, 0, stream>>>(PRED, ENCB, DOTS, b0);
  }

  // softmax loss + accuracy, then normalize
  k_loss<<<(PB * PK * PR * PC + 255) / 256, 256, 0, stream>>>(DOTS, neg_rows, neg_cols, ACC);
  k_final<<<1, 32, 0, stream>>>(ACC, out);
}
